// HeteroGNNLayer_89644557402630
// MI455X (gfx1250) — matrix operand order repaired and checked by reading, not compiled, not run on hardware
//
#include <hip/hip_runtime.h>

#define D 64

typedef __bf16 bf16_t;
typedef bf16_t   v16bf __attribute__((ext_vector_type(16)));
typedef bf16_t   v2bf  __attribute__((ext_vector_type(2)));
typedef float    v8f   __attribute__((ext_vector_type(8)));
typedef float    v2f   __attribute__((ext_vector_type(2)));
typedef unsigned v8u   __attribute__((ext_vector_type(8)));

union frag_u { v8u u; v16bf b; };

// ---------- helpers ----------

// two f32 -> packed bf16x2 in a uint (low = x0); <2xf32> fptrunc maps straight
// onto v_cvt_pk_bf16_f32 with no byte-perm fixups.
__device__ __forceinline__ unsigned pack_bf16(float x0, float x1) {
  v2f f; f[0] = x0; f[1] = x1;
  union { v2bf b; unsigned u; } v;
  v.b = __builtin_convertvector(f, v2bf);
  return v.u;
}

__device__ __forceinline__ v8f wmma_bf16(v16bf a, v16bf b, v8f c) {
  // D(16x16 f32) = A(16x32 bf16) * B(32x16 bf16) + C
  return __builtin_amdgcn_wmma_f32_16x16x32_bf16(false, a, false, b, (short)0, c,
                                                 false, false);
}

// A fragment (16x32) K index for VGPR pair r, lane-half h (ISA 16-bit A layout):
// VGPR0: K=0,1 | 8,9 ; VGPR1: K=2,3 | 10,11 ; ... VGPR4-7: +16.
__device__ __forceinline__ int a_k(int r, int h) {
  return ((r >> 2) << 4) + (h << 3) + ((r & 3) << 1);
}

// Stage a row-major 64x64 f32 matrix into LDS as [n][k/2] packed bf16 pairs.
__device__ __forceinline__ void stage_w(const float* __restrict__ W, unsigned* sW,
                                        int tid, int nthr) {
  for (int i = tid; i < D * D / 2; i += nthr) {
    const int n = i >> 5, k = (i & 31) << 1;           // 32 k-pairs per column
    sW[i] = pack_bf16(W[k * D + n], W[(k + 1) * D + n]);
  }
}

// Build the 8 loop-invariant B fragments (32x16 KxN tiles of the 64x64 weight).
// K order matches the A fragment (a_k): VGPR r holds K = kBase+a_k(r,h), +1.
__device__ __forceinline__ void make_bfrags(const unsigned* sW, int lane, v16bf bf[8]) {
  const int n = lane & 15, h = lane >> 4;
#pragma unroll
  for (int j = 0; j < 4; ++j) {
#pragma unroll
    for (int ks = 0; ks < 2; ++ks) {
      frag_u fu;
#pragma unroll
      for (int r = 0; r < 8; ++r) {
        const int k = ks * 32 + a_k(r, h);
        fu.u[r] = sW[(j * 16 + n) * 32 + (k >> 1)];
      }
      bf[j * 2 + ks] = fu.b;
    }
  }
}

// ---------- kernels ----------

// out[n,:] = feat[n,:] @ W (64x64 row-major); persistent waves, 16 rows per tile.
__global__ void node_linear_kernel(const float* __restrict__ feat,
                                   const float* __restrict__ W,
                                   float* __restrict__ out, int N) {
  __shared__ unsigned sW[D * D / 2];
  stage_w(W, sW, threadIdx.x, blockDim.x);
  __syncthreads();

  const int lane = threadIdx.x & 31, wave = threadIdx.x >> 5;
  const int m = lane & 15, h = lane >> 4;
  v16bf bf[8];
  make_bfrags(sW, lane, bf);

  const int tiles = (N + 15) >> 4;
  const int stride = gridDim.x * (blockDim.x >> 5);
  for (int t = blockIdx.x * (blockDim.x >> 5) + wave; t < tiles; t += stride) {
    const long rowBase = (long)t << 4;
    long rm = rowBase + m; if (rm >= N) rm = N - 1;
    const float* fr = feat + rm * D;

    frag_u a0, a1;
#pragma unroll
    for (int r = 0; r < 8; ++r) {
      const int k = a_k(r, h);
      a0.u[r] = pack_bf16(fr[k],      fr[k + 1]);
      a1.u[r] = pack_bf16(fr[k + 32], fr[k + 33]);
    }

    if (rowBase + 16 <= N) {            // wave-uniform fast path: no store guards
      float* orow = out + (rowBase + 8 * h) * D + m;
#pragma unroll
      for (int j = 0; j < 4; ++j) {
        v8f acc;
#pragma unroll
        for (int r = 0; r < 8; ++r) acc[r] = 0.f;
        acc = wmma_bf16(a0.b, bf[j * 2 + 0], acc);
        acc = wmma_bf16(a1.b, bf[j * 2 + 1], acc);
#pragma unroll
        for (int r = 0; r < 8; ++r) orow[r * D + j * 16] = acc[r];
      }
    } else {                            // tail tile (never taken when 16 | N)
#pragma unroll
      for (int j = 0; j < 4; ++j) {
        v8f acc;
#pragma unroll
        for (int r = 0; r < 8; ++r) acc[r] = 0.f;
        acc = wmma_bf16(a0.b, bf[j * 2 + 0], acc);
        acc = wmma_bf16(a1.b, bf[j * 2 + 1], acc);
#pragma unroll
        for (int r = 0; r < 8; ++r) {
          const long row = rowBase + r + 8 * h;
          if (row < N) out[row * D + j * 16 + m] = acc[r];
        }
      }
    }
  }
}

__global__ void copy_kernel(float* __restrict__ dst, const float* __restrict__ src,
                            long n) {
  const long i = (long)blockIdx.x * blockDim.x + threadIdx.x;
  if (i < n) dst[i] = src[i];
}

__global__ void init_kernel(float* __restrict__ segmax, float* __restrict__ denom,
                            float* __restrict__ agg, int N) {
  const long i = (long)blockIdx.x * blockDim.x + threadIdx.x;
  if (i < (long)N * D) agg[i] = 0.f;
  if (i < N) { segmax[i] = __uint_as_float(0xFF800000u); denom[i] = 0.f; }
}

__global__ void update_kernel(float* __restrict__ feat, const float* __restrict__ agg,
                              const float* __restrict__ denom, int N) {
  const long i = (long)blockIdx.x * blockDim.x + threadIdx.x;
  if (i >= (long)N * D) return;
  const float dn = denom[i >> 6];
  if (dn > 0.f) feat[i] = agg[i] / dn;  // else keep old feature (already in feat)
}

// Persistent waves, 16 edges per tile: hidden = ReLU(pa[src]+pb[dst]+b1);
// msg = hidden@W2 + b2.
// PASS2=false: segmax[dst] = max(segmax[dst], -|msg|/tau)  (float-max of values
//              <= 0 implemented as uint-min of bit patterns)
// PASS2=true : w = exp(logit - segmax[dst]); denom[dst]+=w; agg[dst]+=w*msg
template <bool PASS2>
__global__ void edge_kernel(const float* __restrict__ pa, const float* __restrict__ pb,
                            const float* __restrict__ b1g, const float* __restrict__ W2,
                            const float* __restrict__ b2g,
                            const int* __restrict__ src, const int* __restrict__ dst,
                            int E, const float* __restrict__ tau_p,
                            float* __restrict__ segmax, float* __restrict__ denom,
                            float* __restrict__ agg) {
  __shared__ unsigned sW[D * D / 2];
  __shared__ float sb1[D];
  __shared__ float sb2[D];
  stage_w(W2, sW, threadIdx.x, blockDim.x);
  if (threadIdx.x < D) { sb1[threadIdx.x] = b1g[threadIdx.x];
                         sb2[threadIdx.x] = b2g[threadIdx.x]; }
  __syncthreads();

  const int lane = threadIdx.x & 31, wave = threadIdx.x >> 5;
  const int m = lane & 15, h = lane >> 4;
  const float tau = tau_p[0];

  // loop-invariant register caches: W2 B-fragments, this lane's b1 slice, b2 biases
  v16bf bf[8];
  make_bfrags(sW, lane, bf);
  float b1A[16], b1B[16];
#pragma unroll
  for (int r = 0; r < 8; ++r) {
    const int k = a_k(r, h);
    b1A[2 * r] = sb1[k];      b1A[2 * r + 1] = sb1[k + 1];
    b1B[2 * r] = sb1[k + 32]; b1B[2 * r + 1] = sb1[k + 33];
  }
  float b2v[4];
#pragma unroll
  for (int j = 0; j < 4; ++j) b2v[j] = sb2[j * 16 + m];

  const int tiles = (E + 15) >> 4;
  const int stride = gridDim.x * (blockDim.x >> 5);
  for (int t = blockIdx.x * (blockDim.x >> 5) + wave; t < tiles; t += stride) {
    const long eBase = (long)t << 4;
    long em = eBase + m; if (em >= E) em = E - 1;
    const int s = src[em], d0 = dst[em];
    const float* prA = pa + (long)s * D;
    const float* prB = pb + (long)d0 * D;

    frag_u a0, a1;  // hidden row of edge m, K=0..31 and K=32..63
#pragma unroll
    for (int r = 0; r < 8; ++r) {
      const int k = a_k(r, h);
      const float x0 = fmaxf(prA[k]      + prB[k]      + b1A[2 * r],     0.f);
      const float x1 = fmaxf(prA[k + 1]  + prB[k + 1]  + b1A[2 * r + 1], 0.f);
      const float y0 = fmaxf(prA[k + 32] + prB[k + 32] + b1B[2 * r],     0.f);
      const float y1 = fmaxf(prA[k + 33] + prB[k + 33] + b1B[2 * r + 1], 0.f);
      a0.u[r] = pack_bf16(x0, x1);
      a1.u[r] = pack_bf16(y0, y1);
    }

    v8f c[4];
    float nsq[8];
#pragma unroll
    for (int r = 0; r < 8; ++r) nsq[r] = 0.f;
#pragma unroll
    for (int j = 0; j < 4; ++j) {
      v8f acc;
#pragma unroll
      for (int r = 0; r < 8; ++r) acc[r] = 0.f;
      acc = wmma_bf16(a0.b, bf[j * 2 + 0], acc);
      acc = wmma_bf16(a1.b, bf[j * 2 + 1], acc);
#pragma unroll
      for (int r = 0; r < 8; ++r) {
        const float v = acc[r] + b2v[j];
        acc[r] = v;
        nsq[r] += v * v;  // per-lane partial of row (r + 8h)'s squared norm
      }
      c[j] = acc;
    }
    // butterfly across the 16 lanes of each half -> full row norms in every lane
#pragma unroll
    for (int r = 0; r < 8; ++r) {
      float v = nsq[r];
      v += __shfl_xor(v, 1);
      v += __shfl_xor(v, 2);
      v += __shfl_xor(v, 4);
      v += __shfl_xor(v, 8);
      nsq[r] = v;
    }

    if (!PASS2) {
#pragma unroll
      for (int r = 0; r < 8; ++r) {   // lane m==r of each half writes edge r+8h
        const long er = eBase + r + 8 * h;
        if (m == r && er < E) {
          const int dd = dst[er];
          const float logit = -sqrtf(nsq[r]) / tau;
          atomicMin((unsigned int*)(segmax + dd), __float_as_uint(logit));
        }
      }
    } else {
#pragma unroll
      for (int r = 0; r < 8; ++r) {
        const long er = eBase + r + 8 * h;
        if (er >= E) continue;
        const int dd = dst[er];
        const float logit = -sqrtf(nsq[r]) / tau;
        const float w = __expf(logit - segmax[dd]);
        if (m == 0) atomicAdd(denom + dd, w);       // one lane per half per row
        float* ag = agg + (long)dd * D;
#pragma unroll
        for (int j = 0; j < 4; ++j)
          atomicAdd(ag + j * 16 + m, c[j][r] * w);
      }
    }
  }
}

// ---------- host ----------

static inline int ceil_div_l(long a, int b) { return (int)((a + b - 1) / b); }
static inline int imin(int a, int b) { return a < b ? a : b; }

static void run_stage(const float* srcFeat, int nS, const float* dstFeat, int nD,
                      const int* esrc, const int* edst, int E,
                      const float* W1, const float* b1, const float* W2, const float* b2,
                      const float* tau, float* featUpd,
                      float* pa, float* pb, float* agg, float* segmax, float* denom,
                      hipStream_t stream) {
  const int BLK = 256, WPB = BLK / 32;  // 8 waves/block, 16 rows or edges per tile
  const int gS = imin(ceil_div_l(nS, 16 * WPB), 512);
  const int gD = imin(ceil_div_l(nD, 16 * WPB), 512);
  node_linear_kernel<<<gS, BLK, 0, stream>>>(srcFeat, W1, pa, nS);
  node_linear_kernel<<<gD, BLK, 0, stream>>>(dstFeat, W1 + D * D, pb, nD);
  init_kernel<<<ceil_div_l((long)nD * D, BLK), BLK, 0, stream>>>(segmax, denom, agg, nD);
  const int gE = imin(ceil_div_l(E, 16 * WPB), 1024);
  edge_kernel<false><<<gE, BLK, 0, stream>>>(pa, pb, b1, W2, b2, esrc, edst, E, tau,
                                             segmax, nullptr, nullptr);
  edge_kernel<true><<<gE, BLK, 0, stream>>>(pa, pb, b1, W2, b2, esrc, edst, E, tau,
                                            segmax, denom, agg);
  update_kernel<<<ceil_div_l((long)nD * D, BLK), BLK, 0, stream>>>(featUpd, agg, denom,
                                                                   nD);
}

extern "C" void kernel_launch(void* const* d_in, const int* in_sizes, int n_in,
                              void* d_out, int out_size, void* d_ws, size_t ws_size,
                              hipStream_t stream) {
  (void)n_in; (void)out_size; (void)ws_size;
  const float* tile_feat = (const float*)d_in[0];
  const float* rr_feat   = (const float*)d_in[1];
  const float* tau       = (const float*)d_in[2];
  const int* t2t_src = (const int*)d_in[3];
  const int* t2t_dst = (const int*)d_in[4];
  const int* r2t_src = (const int*)d_in[5];
  const int* r2t_dst = (const int*)d_in[6];
  const int* t2r_src = (const int*)d_in[7];
  const int* t2r_dst = (const int*)d_in[8];
  const float* W1_tt = (const float*)d_in[9];  const float* b1_tt = (const float*)d_in[10];
  const float* W2_tt = (const float*)d_in[11]; const float* b2_tt = (const float*)d_in[12];
  const float* W1_rt = (const float*)d_in[13]; const float* b1_rt = (const float*)d_in[14];
  const float* W2_rt = (const float*)d_in[15]; const float* b2_rt = (const float*)d_in[16];
  const float* W1_tr = (const float*)d_in[17]; const float* b1_tr = (const float*)d_in[18];
  const float* W2_tr = (const float*)d_in[19]; const float* b2_tr = (const float*)d_in[20];

  const int nT = in_sizes[0] / D;
  const int nR = in_sizes[1] / D;
  const int E_tt = in_sizes[3], E_rt = in_sizes[5], E_tr = in_sizes[7];
  const int nMax = nT > nR ? nT : nR;

  // d_out doubles as the working feature tables: [tile (nT,64) | rr (nR,64)]
  float* outT = (float*)d_out;
  float* outR = outT + (long)nT * D;

  // workspace: 3 node-sized f32 tables + 2 per-node scalars (~78 MB @ n=100k)
  float* pa     = (float*)d_ws;
  float* pb     = pa + (long)nMax * D;
  float* agg    = pb + (long)nMax * D;
  float* segmax = agg + (long)nMax * D;
  float* denom  = segmax + nMax;

  const int BLK = 256;
  copy_kernel<<<ceil_div_l((long)nT * D, BLK), BLK, 0, stream>>>(outT, tile_feat,
                                                                 (long)nT * D);
  copy_kernel<<<ceil_div_l((long)nR * D, BLK), BLK, 0, stream>>>(outR, rr_feat,
                                                                 (long)nR * D);

  // stage 1: tile -> tile (updates outT)
  run_stage(outT, nT, outT, nT, t2t_src, t2t_dst, E_tt,
            W1_tt, b1_tt, W2_tt, b2_tt, tau, outT,
            pa, pb, agg, segmax, denom, stream);
  // stage 2: rr -> tile (src = original rr, dst = once-updated tile; updates outT)
  run_stage(rr_feat, nR, outT, nT, r2t_src, r2t_dst, E_rt,
            W1_rt, b1_rt, W2_rt, b2_rt, tau, outT,
            pa, pb, agg, segmax, denom, stream);
  // stage 3: tile -> rr (src = twice-updated tile; updates outR)
  run_stage(outT, nT, outR, nR, t2r_src, t2r_dst, E_tr,
            W1_tr, b1_tr, W2_tr, b2_tr, tau, outR,
            pa, pb, agg, segmax, denom, stream);
}
